// satImgqModel_32512902430810
// MI455X (gfx1250) — matrix operand order repaired and checked by reading, not compiled, not run on hardware
//
#include <hip/hip_runtime.h>
#include <hip/hip_fp16.h>
#include <math.h>

typedef _Float16 f16;
typedef __attribute__((ext_vector_type(16))) _Float16 v16h;
typedef __attribute__((ext_vector_type(8)))  _Float16 v8h;
typedef __attribute__((ext_vector_type(8)))  float    v8f;
typedef int v4i_vs __attribute__((vector_size(16)));   // matches builtin param type

#define NT 4           // 16-channel output tiles per wave (4 WMMAs per K-step, A reused)
#define WAVES_PER_BLK 8  // 256 threads; 8 pixel tiles per block share one LDS B stage

#if __has_builtin(__builtin_amdgcn_global_load_async_to_lds_b128) && \
    __has_builtin(__builtin_amdgcn_s_wait_asynccnt)
#define HAVE_ASYNC_LDS 1
#else
#define HAVE_ASYNC_LDS 0
#endif

// ---------------- prep kernels ----------------

__global__ void bn_fold_kernel(const float* __restrict__ g, const float* __restrict__ b,
                               const float* __restrict__ m, const float* __restrict__ v,
                               float* __restrict__ scale, float* __restrict__ bias, int C) {
  int i = blockIdx.x * blockDim.x + threadIdx.x;
  if (i < C) {
    float s = g[i] * rsqrtf(v[i] + 1e-5f);
    scale[i] = s;
    bias[i]  = b[i] - m[i] * s;
  }
}

__global__ void fill_sb_kernel(const float* __restrict__ bsrc, float* __restrict__ scale,
                               float* __restrict__ bias, int C) {
  int i = blockIdx.x * blockDim.x + threadIdx.x;
  if (i < C) { scale[i] = 1.0f; bias[i] = bsrc[i]; }
}

// Repack f32 OIHW weights into the per-lane B-operand register image:
// packed[((kt*ntiles + nt)*32 + lane)*16 + i]  holds  B(k = kt*32 + (lane>=16?8:0) + (i&7) + (i>>3)*16,
//                                                       n = nt*16 + (lane&15))
// K ordering is (kh, kw, c) with c innermost; zero-padded for k>=K or n>=O.
__global__ void pack_weights_kernel(const float* __restrict__ w, f16* __restrict__ packed,
                                    int O, int I, int KH, int KW, int ktiles, int ntiles) {
  int t = blockIdx.x * blockDim.x + threadIdx.x;
  int total = ktiles * ntiles * 32;
  if (t >= total) return;
  int lane = t & 31;
  int tile = t >> 5;
  int nt = tile % ntiles;
  int kt = tile / ntiles;
  int n  = nt * 16 + (lane & 15);
  int kb = kt * 32 + (lane >> 4) * 8;
  int K  = KH * KW * I;
  f16* dst = packed + (size_t)t * 16;
#pragma unroll
  for (int i = 0; i < 16; ++i) {
    int k = kb + (i & 7) + (i >> 3) * 16;
    float val = 0.0f;
    if (n < O && k < K) {
      int kh = k / (KW * I);
      int r  = k % (KW * I);
      int kw = r / I;
      int c  = r % I;
      val = w[(((size_t)n * I + c) * KH + kh) * KW + kw];
    }
    dst[i] = (f16)val;
  }
}

__global__ void nchw2nhwc_kernel(const float* __restrict__ in, f16* __restrict__ out,
                                 int N, int C, int H, int W) {
  long long idx = (long long)blockIdx.x * blockDim.x + threadIdx.x;
  long long total = (long long)N * H * W * C;
  if (idx >= total) return;
  int c = (int)(idx % C);
  long long r = idx / C;
  int w = (int)(r % W); r /= W;
  int h = (int)(r % H); r /= H;
  int n = (int)r;
  out[idx] = (f16)in[(((long long)n * C + c) * H + h) * W + w];
}

__global__ void maxpool_kernel(const f16* __restrict__ in, f16* __restrict__ out,
                               int N, int H, int W, int C, int OH, int OW) {
  long long idx = (long long)blockIdx.x * blockDim.x + threadIdx.x;
  long long total = (long long)N * OH * OW * C;
  if (idx >= total) return;
  int c = (int)(idx % C);
  long long r = idx / C;
  int ow = (int)(r % OW); r /= OW;
  int oh = (int)(r % OH); r /= OH;
  int n = (int)r;
  float m = -1e30f;
  for (int kh = 0; kh < 3; ++kh)
    for (int kw = 0; kw < 3; ++kw) {
      int ih = oh * 2 - 1 + kh, iw = ow * 2 - 1 + kw;
      if (ih >= 0 && ih < H && iw >= 0 && iw < W)
        m = fmaxf(m, (float)in[(((size_t)n * H + ih) * W + iw) * C + c]);
    }
  out[idx] = (f16)m;
}

__global__ void avgpool_kernel(const f16* __restrict__ in, f16* __restrict__ out,
                               int N, int HW, int C) {
  int idx = blockIdx.x * blockDim.x + threadIdx.x;
  if (idx >= N * C) return;
  int n = idx / C, c = idx % C;
  const f16* p = in + (size_t)n * HW * C + c;
  float s = 0.f;
  for (int i = 0; i < HW; ++i) s += (float)p[(size_t)i * C];
  out[idx] = (f16)(s / (float)HW);
}

// ---------------- implicit-GEMM conv via WMMA + LDS-staged B ----------------
// in:  NHWC f16, channel stride = Cin
// out: NHWC f16, channel stride = CoutPad (lanes with n>=Cout write 0)
// Requires ntiles % NT == 0 (all CoutPad are multiples of 64).
// Block = 8 waves: all share one channel group (NT tiles), take 8 consecutive
// pixel tiles. B tiles double-buffered in LDS: one b128 async load per thread
// per K-step, overlapped with the current step's WMMAs.
__global__ void conv_wmma_kernel(const f16* __restrict__ in, const f16* __restrict__ pw,
                                 const float* __restrict__ scale, const float* __restrict__ bias,
                                 const f16* __restrict__ residual, f16* __restrict__ out,
                                 int N, int H, int W, int Cin,
                                 int OH, int OW, int Cout, int CoutPad,
                                 int KH, int KW, int stride, int pad, int relu,
                                 int ktiles, int ntiles) {
  __shared__ alignas(32) f16 smem[2][NT * 32 * 16];  // 2 x 4KB

  const int tid  = threadIdx.x;
  const int lane = tid & 31;
  const int wv   = tid >> 5;
  const int P = N * OH * OW;
  const int ptiles  = (P + 15) >> 4;
  const int ngroups = ntiles / NT;
  const int ng = blockIdx.x % ngroups;
  const int pt = (blockIdx.x / ngroups) * WAVES_PER_BLK + wv;  // may be >= ptiles (masked)
  const int ngBase = ng * NT;

  const int m = lane & 15;
  const int halfsel = lane >> 4;
  const int p = pt * 16 + m;
  const bool pvalid = p < P;
  const int pp = pvalid ? p : 0;
  const int ow   = pp % OW;
  const int tmp  = pp / OW;
  const int oh   = tmp % OH;
  const int nimg = tmp / OH;
  const int KWI  = KW * Cin;
  const int Ktot = KH * KWI;
  const bool fast = (Cin & 31) == 0;

  // stage B tiles for K-step kt into LDS buffer buf (one 16B chunk per thread)
  auto stage = [&](int kt, int buf) {
    const f16* src = pw + ((size_t)kt * ntiles + ngBase) * 512 + (size_t)tid * 8;
    f16* dst = &smem[buf][tid * 8];
#if HAVE_ASYNC_LDS
    __builtin_amdgcn_global_load_async_to_lds_b128(
        (__attribute__((address_space(1))) v4i_vs*)src,
        (__attribute__((address_space(3))) v4i_vs*)dst, 0, 0);
#else
    *(v8h*)dst = *(const v8h*)src;
#endif
  };

  v8f acc[NT] = {};
  int khc = 0, kwc = 0, c0c = 0;  // incremental (kh,kw,c0) decode for the fast path

  stage(0, 0);

  for (int kt = 0; kt < ktiles; ++kt) {
#if HAVE_ASYNC_LDS
    __builtin_amdgcn_s_wait_asynccnt(0);   // stage(kt) complete, LDS visible
#endif
    __syncthreads();                       // everyone done reading buf[(kt+1)&1] at kt-1
    if (kt + 1 < ktiles) stage(kt + 1, (kt + 1) & 1);

    // ---- build A operand (16 halves per lane) ----
    v16h a;
    if (fast) {
      const int ih = oh * stride - pad + khc;
      const int iw = ow * stride - pad + kwc;
      v8h lo = {}; v8h hi = {};
      if (pvalid && ih >= 0 && ih < H && iw >= 0 && iw < W) {
        const f16* src = in + ((((size_t)nimg * H + ih) * W + iw) * Cin + c0c + halfsel * 8);
        lo = *(const v8h*)src;          // K = c0 + sel*8 + 0..7
        hi = *(const v8h*)(src + 16);   // K = c0 + 16 + sel*8 + 0..7
      }
#pragma unroll
      for (int i = 0; i < 8; ++i) { a[i] = lo[i]; a[i + 8] = hi[i]; }
    } else {
      const int kbase = kt * 32;
#pragma unroll
      for (int i = 0; i < 16; ++i) {
        int k = kbase + ((i >> 3) * 16) + halfsel * 8 + (i & 7);
        f16 val = (f16)0.0f;
        if (pvalid && k < Ktot) {
          int kh = k / KWI; int r = k % KWI; int kw = r / Cin; int c = r % Cin;
          int ih = oh * stride - pad + kh;
          int iw = ow * stride - pad + kw;
          if (ih >= 0 && ih < H && iw >= 0 && iw < W)
            val = in[(((size_t)nimg * H + ih) * W + iw) * Cin + c];
        }
        a[i] = val;
      }
    }

    // ---- B from LDS, 4 WMMAs ----
    const f16* sb = smem[kt & 1];
#pragma unroll
    for (int j = 0; j < NT; ++j) {
      const v16h b = *(const v16h*)(sb + ((size_t)j * 32 + lane) * 16);
      acc[j] = __builtin_amdgcn_wmma_f32_16x16x32_f16(
          false, a, false, b, (short)0, acc[j], false, false);
    }

    // advance fast-path (kh,kw,c0) by 32 K (Cin % 32 == 0 on this path)
    c0c += 32;
    if (c0c >= Cin) { c0c = 0; if (++kwc == KW) { kwc = 0; ++khc; } }
  }

  // fused BN / bias / residual / ReLU epilogue. D layout: lane holds column
  // n = nt*16 + (lane&15); VGPR r holds row m = r + 8*(lane>=16).
#pragma unroll
  for (int j = 0; j < NT; ++j) {
    int n = (ngBase + j) * 16 + (lane & 15);
    float sc = (n < Cout) ? scale[n] : 0.0f;
    float bs = (n < Cout) ? bias[n]  : 0.0f;
#pragma unroll
    for (int r = 0; r < 8; ++r) {
      int mr = r + 8 * halfsel;
      int p2 = pt * 16 + mr;
      if (p2 < P) {
        float v = acc[j][r] * sc + bs;
        size_t oidx = (size_t)p2 * CoutPad + n;
        if (residual) v += (float)residual[oidx];
        if (relu) v = fmaxf(v, 0.0f);
        out[oidx] = (f16)v;
      }
    }
  }
}

// ---------------- precision tail: lin(1000->16), bn16, gelu, quantum, log-softmax ----------------
__global__ void tail_kernel(const f16* __restrict__ act /*64 x 1024, C<1000 valid*/,
                            const float* __restrict__ lin_w, const float* __restrict__ lin_b,
                            const float* __restrict__ bn_b, const float* __restrict__ bn_g,
                            const float* __restrict__ bn_m, const float* __restrict__ bn_v,
                            const float* __restrict__ qw, float* __restrict__ out) {
  int smp = blockIdx.x;
  int t = threadIdx.x;
  __shared__ float feats[16];
  if (t < 16) {
    float acc = lin_b[t];
    const f16* a = act + (size_t)smp * 1024;
    for (int i = 0; i < 1000; ++i) acc += (float)a[i] * lin_w[t * 1000 + i];
    float sc = bn_g[t] * rsqrtf(bn_v[t] + 1e-5f);
    float x = (acc - bn_m[t]) * sc + bn_b[t];
    x = 0.5f * x * (1.0f + erff(x * 0.7071067811865475f));  // exact gelu
    feats[t] = x;
  }
  __syncthreads();
  if (t == 0) {
    float nrm = 0.f;
    for (int i = 0; i < 16; ++i) nrm += feats[i] * feats[i];
    nrm = sqrtf(nrm);
    float sr[16], si[16];
    for (int i = 0; i < 16; ++i) { sr[i] = feats[i] / nrm; si[i] = 0.f; }
    const int pc[4] = {0, 1, 2, 3}, ptg[4] = {1, 2, 3, 0};
    for (int l = 0; l < 6; ++l) {
      for (int w = 0; w < 4; ++w) {
        float th = qw[l * 4 + w] * 0.5f;
        float c = cosf(th), s = sinf(th);   // U = [[c, -i s], [-i s, c]]
        int bit = 1 << (3 - w);
        for (int i = 0; i < 16; ++i) {
          if (i & bit) continue;
          int j = i | bit;
          float r0 = sr[i], i0 = si[i], r1 = sr[j], i1 = si[j];
          sr[i] = c * r0 + s * i1;  si[i] = c * i0 - s * r1;
          sr[j] = s * i0 + c * r1;  si[j] = -s * r0 + c * i1;
        }
      }
      for (int pi = 0; pi < 4; ++pi) {
        int cb = 1 << (3 - pc[pi]), tb = 1 << (3 - ptg[pi]);
        float tr[16], ti[16];
        for (int i = 0; i < 16; ++i) {
          int src = (i & cb) ? (i ^ tb) : i;
          tr[i] = sr[src]; ti[i] = si[src];
        }
        for (int i = 0; i < 16; ++i) { sr[i] = tr[i]; si[i] = ti[i]; }
      }
    }
    float ev[4];
    for (int w = 0; w < 4; ++w) {
      float e = 0.f; int bit = 1 << (3 - w);
      for (int i = 0; i < 16; ++i) {
        float pr = sr[i] * sr[i] + si[i] * si[i];
        e += (i & bit) ? -pr : pr;
      }
      ev[w] = e;
    }
    float mx = fmaxf(fmaxf(ev[0], ev[1]), fmaxf(ev[2], ev[3]));
    float sum = 0.f;
    for (int i = 0; i < 4; ++i) sum += expf(ev[i] - mx);
    float lse = logf(sum) + mx;
    for (int i = 0; i < 4; ++i) out[smp * 4 + i] = ev[i] - lse;
  }
}

// ---------------- host orchestration ----------------

static inline int cdiv(int a, int b) { return (a + b - 1) / b; }

static void launch_conv(hipStream_t stream, const f16* in, const f16* packed,
                        const float* scale, const float* bias, const f16* residual, f16* out,
                        int N, int H, int W, int Cin, int OH, int OW, int Cout, int CoutPad,
                        int KH, int KW, int stride, int pad, int relu, int ktiles, int ntiles) {
  int P = N * OH * OW;
  int ptiles = (P + 15) / 16;
  int ngroups = ntiles / NT;                       // ntiles % NT == 0 by construction
  int ptgroups = cdiv(ptiles, WAVES_PER_BLK);
  int grid = ptgroups * ngroups;
  conv_wmma_kernel<<<grid, 256, 0, stream>>>(in, packed, scale, bias, residual, out,
                                             N, H, W, Cin, OH, OW, Cout, CoutPad,
                                             KH, KW, stride, pad, relu, ktiles, ntiles);
}

extern "C" void kernel_launch(void* const* d_in, const int* in_sizes, int n_in,
                              void* d_out, int out_size, void* d_ws, size_t ws_size,
                              hipStream_t stream) {
  (void)in_sizes; (void)n_in; (void)out_size; (void)ws_size;
  char* base = (char*)d_ws;
  size_t off = 0;
  auto alloc = [&](size_t bytes) -> char* {
    char* p = base + off;
    off += (bytes + 255) & ~(size_t)255;
    return p;
  };
  auto F = [&](int i) -> const float* { return (const float*)d_in[i]; };
  const int B = 64;

  // ---- leaf indices (jax pytree order: dicts sorted by key) ----
  // 0 image | 1..4 bn1{b,g,m,v} | 5..8 bn16 | 9 conv1 | 10 fc_b | 11 fc_w |
  // layers: per block bn1(4) bn2(4) c1 c2 [dbn(4) dw] | 107 lin_b | 108 lin_w | 109 qweights

  struct ConvW { int leaf, O, I, KH, KW, CoutPad; f16* packed; int ktiles, ntiles; };
  ConvW convs[] = {
    {9,   64,   3, 7, 7,   64},                              // 0: conv1
    {20,  64,  64, 3, 3,   64}, {21,  64,  64, 3, 3,   64},  // l0b0 c1,c2
    {30,  64,  64, 3, 3,   64}, {31,  64,  64, 3, 3,   64},  // l0b1
    {40, 128,  64, 3, 3,  128}, {41, 128, 128, 3, 3,  128}, {46, 128,  64, 1, 1, 128},  // l1b0 c1,c2,dw
    {55, 128, 128, 3, 3,  128}, {56, 128, 128, 3, 3,  128},  // l1b1
    {65, 256, 128, 3, 3,  256}, {66, 256, 256, 3, 3,  256}, {71, 256, 128, 1, 1, 256},  // l2b0
    {80, 256, 256, 3, 3,  256}, {81, 256, 256, 3, 3,  256},  // l2b1
    {90, 512, 256, 3, 3,  512}, {91, 512, 512, 3, 3,  512}, {96, 512, 256, 1, 1, 512},  // l3b0
    {105,512, 512, 3, 3,  512}, {106,512, 512, 3, 3,  512},  // l3b1
    {11, 1000, 512, 1, 1, 1024},                             // 20: fc 512->1000 (padded to 1024)
  };
  const int NCONV = (int)(sizeof(convs) / sizeof(convs[0]));
  for (int i = 0; i < NCONV; ++i) {
    ConvW& c = convs[i];
    c.ktiles = cdiv(c.KH * c.KW * c.I, 32);
    c.ntiles = c.CoutPad / 16;
    c.packed = (f16*)alloc((size_t)c.ktiles * c.ntiles * 32 * 16 * sizeof(f16));
    int threads = c.ktiles * c.ntiles * 32;
    pack_weights_kernel<<<cdiv(threads, 256), 256, 0, stream>>>(
        F(c.leaf), c.packed, c.O, c.I, c.KH, c.KW, c.ktiles, c.ntiles);
  }

  struct BN { int leaf, C; float *scale, *bias; };
  BN bns[] = {
    {1, 64},                                   // top bn1
    {12, 64}, {16, 64},                        // l0b0 bn1,bn2
    {22, 64}, {26, 64},                        // l0b1
    {32, 128}, {36, 128}, {42, 128},           // l1b0 bn1,bn2,dbn
    {47, 128}, {51, 128},                      // l1b1
    {57, 256}, {61, 256}, {67, 256},           // l2b0
    {72, 256}, {76, 256},                      // l2b1
    {82, 512}, {86, 512}, {92, 512},           // l3b0
    {97, 512}, {101, 512},                     // l3b1
  };
  const int NBN = (int)(sizeof(bns) / sizeof(bns[0]));
  for (int i = 0; i < NBN; ++i) {
    BN& bn = bns[i];
    bn.scale = (float*)alloc(bn.C * sizeof(float));
    bn.bias  = (float*)alloc(bn.C * sizeof(float));
    // leaves stored in order b,g,m,v
    bn_fold_kernel<<<cdiv(bn.C, 256), 256, 0, stream>>>(
        F(bn.leaf + 1), F(bn.leaf + 0), F(bn.leaf + 2), F(bn.leaf + 3),
        bn.scale, bn.bias, bn.C);
  }
  float* fc_scale = (float*)alloc(1000 * sizeof(float));
  float* fc_bias  = (float*)alloc(1000 * sizeof(float));
  fill_sb_kernel<<<cdiv(1000, 256), 256, 0, stream>>>(F(10), fc_scale, fc_bias, 1000);

  // ---- activation buffers (all NHWC f16) ----
  f16* img16 = (f16*)alloc((size_t)B * 224 * 224 * 3 * sizeof(f16));
  f16* a0    = (f16*)alloc((size_t)B * 112 * 112 * 64 * sizeof(f16));
  size_t pb  = (size_t)B * 56 * 56 * 64 * sizeof(f16);
  f16* pool[3] = { (f16*)alloc(pb), (f16*)alloc(pb), (f16*)alloc(pb) };
  f16* scbuf = (f16*)alloc((size_t)B * 28 * 28 * 128 * sizeof(f16));
  f16* gap   = (f16*)alloc((size_t)B * 512 * sizeof(f16));
  f16* fcact = (f16*)alloc((size_t)B * 1024 * sizeof(f16));

  { // image f32 NCHW -> f16 NHWC
    int total = B * 224 * 224 * 3;
    nchw2nhwc_kernel<<<cdiv(total, 256), 256, 0, stream>>>(F(0), img16, B, 3, 224, 224);
  }
  // conv1 7x7 s2 + bn + relu (scalar-A path, Cin=3)
  launch_conv(stream, img16, convs[0].packed, bns[0].scale, bns[0].bias, nullptr, a0,
              B, 224, 224, 3, 112, 112, 64, 64, 7, 7, 2, 3, 1,
              convs[0].ktiles, convs[0].ntiles);
  { // maxpool 3x3 s2 pad1: 112 -> 56
    long long total = (long long)B * 56 * 56 * 64;
    maxpool_kernel<<<(int)((total + 255) / 256), 256, 0, stream>>>(
        a0, pool[0], B, 112, 112, 64, 56, 56);
  }

  // ---- residual stages ----
  int H = 56, Cin = 64;
  int xi = 0; f16* x = pool[0];
  int ci = 1, bi2 = 1;
  const int chans[4] = {64, 128, 256, 512};
  for (int li = 0; li < 4; ++li) {
    int oc = chans[li];
    for (int blk = 0; blk < 2; ++blk) {
      int stride = (li > 0 && blk == 0) ? 2 : 1;
      int OHW = H / stride;
      bool down = (stride != 1 || Cin != oc);
      int yi1 = (xi + 1) % 3, yi2 = (xi + 2) % 3;
      f16* y1 = pool[yi1]; f16* y2 = pool[yi2];
      ConvW& c1 = convs[ci++];
      BN& b1 = bns[bi2++];
      BN& b2 = bns[bi2++];
      launch_conv(stream, x, c1.packed, b1.scale, b1.bias, nullptr, y1,
                  B, H, H, Cin, OHW, OHW, oc, oc, 3, 3, stride, 1, 1,
                  c1.ktiles, c1.ntiles);
      ConvW& c2 = convs[ci++];
      const f16* res;
      if (down) {
        ConvW& cd = convs[ci++];
        BN& bd = bns[bi2++];
        launch_conv(stream, x, cd.packed, bd.scale, bd.bias, nullptr, scbuf,
                    B, H, H, Cin, OHW, OHW, oc, oc, 1, 1, stride, 0, 0,
                    cd.ktiles, cd.ntiles);
        res = scbuf;
      } else {
        res = x;
      }
      launch_conv(stream, y1, c2.packed, b2.scale, b2.bias, res, y2,
                  B, OHW, OHW, oc, OHW, OHW, oc, oc, 3, 3, 1, 1, 1,
                  c2.ktiles, c2.ntiles);
      x = y2; xi = yi2;
      H = OHW; Cin = oc;
    }
  }

  { // global average pool 7x7x512 -> 64x512
    int total = B * 512;
    avgpool_kernel<<<cdiv(total, 256), 256, 0, stream>>>(x, gap, B, 49, 512);
  }
  { // fc 512->1000 as 1x1 conv on 64 one-pixel images (WMMA); pads 1000..1023 with zeros
    ConvW& cf = convs[20];
    launch_conv(stream, gap, cf.packed, fc_scale, fc_bias, nullptr, fcact,
                B, 1, 1, 512, 1, 1, 1000, 1024, 1, 1, 1, 0, 0,
                cf.ktiles, cf.ntiles);
  }
  // lin 1000->16, bn16, exact gelu, quantum circuit, log-softmax
  tail_kernel<<<B, 32, 0, stream>>>(fcact, F(108), F(107),
                                    F(5), F(6), F(7), F(8),
                                    F(109), (float*)d_out);
}
